// LSTM_52338471469521
// MI455X (gfx1250) — matrix operand order repaired and checked by reading, not compiled, not run on hardware
//
#include <hip/hip_runtime.h>

// MI455X / gfx1250: wave32, WMMA 16x16x32 f16->f32, async global->LDS copies.
typedef __attribute__((ext_vector_type(16))) _Float16 v16h;
typedef __attribute__((ext_vector_type(8)))  float    v8f;

#define B_    32
#define H_    512
#define I_    512
#define T_    32
#define A_    16
#define KTOT  1536          // [x_t(512) | atten(512) | h(512)]
#define NGATE 2048          // 4*H
#define KC_N  48            // KTOT/32
#define NWG   16            // persistent workgroups
#define LDSTR 1552          // LDS act row stride in halves (3104B: 32B-aligned, low-conflict)

// ---------------------------------------------------------------------------
// Pack Wc = [W_ih | W_hh] (2048 x 1536) as f16 per lane in contiguous k order (k = 16*(lane/16) + e) for
// V_WMMA_F32_16X16X32_F16 (B loads use the same k order), and fold the two bias vectors.
// ---------------------------------------------------------------------------
__global__ void pack_weights(const float* __restrict__ Wih,
                             const float* __restrict__ Whh,
                             const float* __restrict__ bih,
                             const float* __restrict__ bhh,
                             _Float16* __restrict__ wpack,
                             float* __restrict__ biasc) {
  int i = blockIdx.x * 256 + threadIdx.x;           // i in [0, 2048*1536)
  if (i < NGATE) biasc[i] = bih[i] + bhh[i];
  int e     = i & 15;
  int lane  = (i >> 4) & 31;
  int chunk = i >> 9;                               // mt*KC_N + kc
  int kc    = chunk % KC_N;
  int mt    = chunk / KC_N;
  if (mt >= NGATE / 16) return;
  int k;
  if (lane < 16) k = kc * 32 + e;
  else           k = kc * 32 + 16 + e;
  int r = mt * 16 + (lane & 15);
  float w = (k < 1024) ? Wih[r * 1024 + k] : Whh[r * 512 + (k - 1024)];
  wpack[i] = (_Float16)w;
}

// Branch-free hardware-rate nonlinearities (v_exp_f32 / v_rcp_f32).
__device__ __forceinline__ float sig_(float x) {
  return __builtin_amdgcn_rcpf(1.f + __builtin_amdgcn_exp2f(x * -1.4426950408889634f));
}
__device__ __forceinline__ float tanh_(float x) {
  return 2.f * __builtin_amdgcn_rcpf(1.f + __builtin_amdgcn_exp2f(x * -2.8853900817779268f)) - 1.f;
}

// Device-wide barrier across the 16 persistent WGs.
__device__ __forceinline__ void gbar(unsigned* ctr, unsigned target) {
  __builtin_amdgcn_fence(__ATOMIC_RELEASE, "agent");
  __syncthreads();
  if (threadIdx.x == 0) {
    atomicAdd(ctr, 1u);
    while (atomicAdd(ctr, 0u) < target) __builtin_amdgcn_s_sleep(1);
  }
  __syncthreads();
  __builtin_amdgcn_fence(__ATOMIC_ACQUIRE, "agent");
}

__global__ __launch_bounds__(256)
void lstm_persistent(const float* __restrict__ x_in,
                     const float* __restrict__ h0,
                     const float* __restrict__ c0,
                     const float* __restrict__ out0,
                     const float* __restrict__ biasc,
                     const _Float16* __restrict__ wpack,
                     _Float16* __restrict__ act,
                     unsigned* __restrict__ ctr,
                     float* __restrict__ d_out) {
  const int g    = blockIdx.x;        // 0..15
  const int tid  = threadIdx.x;
  const int w    = tid >> 5;          // wave 0..7
  const int lane = tid & 31;

  __shared__ _Float16 act_lds[32][LDSTR];   // full activation mirror, ~97KB
  __shared__ float h_lds[32][33];
  __shared__ float c_lds[32][33];
  __shared__ float att_lds[32][33];
  __shared__ float gate_lds[4][32][33];

  float* buf_out = d_out;                       // (B, A, H) ring buffer
  float* h_out   = d_out + (size_t)B_ * A_ * H_;
  float* c_out   = h_out + (size_t)B_ * H_;

  // ---- loop-invariant per-thread (b, r) mapping and hoisted state -------
  int bq[4], rq[4];
  const float* xp[4];         // x stream base: + t*I_ per step
  _Float16*    ap[4];         // act publish base (x segment; +512/+1024 for att/h)
  float*       rp[4];         // ring base: + slot*H_ per step
  float bI[4], bF[4], bG[4], bO[4];
  #pragma unroll
  for (int q = 0; q < 4; ++q) {
    int idx = tid + 256 * q;
    int b = idx >> 5, r = idx & 31;
    bq[q] = b; rq[q] = r;
    int col = g * 32 + r;
    xp[q] = x_in + (size_t)b * T_ * I_ + col;
    ap[q] = act + (size_t)b * KTOT + col;
    rp[q] = buf_out + (size_t)b * A_ * H_ + col;
    bI[q] = biasc[col];
    bF[q] = biasc[512 + col];
    bG[q] = biasc[1024 + col];
    bO[q] = biasc[1536 + col];
    // init persistent state
    h_lds[r][b]   = h0[b * H_ + col];
    c_lds[r][b]   = c0[b * H_ + col];
    att_lds[r][b] = out0[b * H_ + col];   // buf0 = zeros except slot 15 = out
  }
  for (int q = 0; q < 64; ++q) {
    int idx = tid + 256 * q;
    int col = idx & 31;
    int a   = (idx >> 5) & 15;
    int b   = idx >> 9;
    float v = (a == 15) ? out0[b * H_ + g * 32 + col] : 0.f;
    buf_out[((size_t)b * A_ + a) * H_ + g * 32 + col] = v;
  }
  __syncthreads();

  const int gt        = w >> 1;
  const int rloc_base = (w & 1) * 16;
  const int mt_glob   = gt * 32 + g * 2 + (w & 1);
  const _Float16* wbase = wpack + ((size_t)mt_glob * KC_N) * 512 + (size_t)lane * 16;
  const int colB = lane & 15;
  const int koff = (lane >> 4) * 16;
  const _Float16* pB0 = &act_lds[colB][koff];        // batch 0..15
  const _Float16* pB1 = &act_lds[16 + colB][koff];   // batch 16..31
  const unsigned lds_act_base = (unsigned)(uintptr_t)&act_lds[0][0];

  unsigned ep = 0;
  for (int t = 0; t < T_; ++t) {
    // ---- publish this step's activation slice to global (f16) ----------
    #pragma unroll
    for (int q = 0; q < 4; ++q) {
      int b = bq[q], r = rq[q];
      ap[q][0]    = (_Float16)xp[q][(size_t)t * I_];
      ap[q][512]  = (_Float16)att_lds[r][b];
      ap[q][1024] = (_Float16)h_lds[r][b];
      (void)b;
    }
    ep++; gbar(ctr, NWG * ep);        // act fully written by all WGs

    // ---- async-copy the full 96KB act matrix into LDS (CDNA5 path) -----
    // 32 rows x 3072B, 16B per lane-transfer: 6144 transfers / 256 threads.
    #pragma unroll
    for (int q = 0; q < 24; ++q) {
      int tau = tid + 256 * q;
      int row = tau / 192;
      int off = tau % 192;                       // 16B units within row
      unsigned dst = lds_act_base + (unsigned)(row * (LDSTR * 2) + off * 16);
      unsigned long long src =
          (unsigned long long)(uintptr_t)((const char*)act + row * (KTOT * 2) + off * 16);
      asm volatile("global_load_async_to_lds_b128 %0, %1, off"
                   :: "v"(dst), "v"(src) : "memory");
    }
    asm volatile("s_wait_asynccnt 0x0" ::: "memory");
    __syncthreads();                  // all waves' async writes landed in LDS

    // ---- GEMM: 16 gate rows x 32 batch per wave, K = 1536 --------------
    v8f acc0 = {}; v8f acc1 = {};
    #pragma unroll 2
    for (int kc = 0; kc < KC_N; ++kc) {
      v16h av  = *(const v16h*)(wbase + (size_t)kc * 512);   // A: prepacked, L2
      v16h b0v = *(const v16h*)(pB0 + kc * 32);              // B: LDS
      v16h b1v = *(const v16h*)(pB1 + kc * 32);
      acc0 = __builtin_amdgcn_wmma_f32_16x16x32_f16(false, av, false, b0v,
                                                    (short)0, acc0, false, false);
      acc1 = __builtin_amdgcn_wmma_f32_16x16x32_f16(false, av, false, b1v,
                                                    (short)0, acc1, false, false);
    }
    {
      int mhi = (lane >> 4) * 8;
      int n   = lane & 15;
      #pragma unroll
      for (int j = 0; j < 8; ++j) {
        gate_lds[gt][rloc_base + mhi + j][n]      = acc0[j];
        gate_lds[gt][rloc_base + mhi + j][16 + n] = acc1[j];
      }
    }
    __syncthreads();

    // ---- LSTM cell + ring/attention-sum update -------------------------
    int slot = t & 15;
    #pragma unroll
    for (int q = 0; q < 4; ++q) {
      int b = bq[q], r = rq[q];
      float iv = gate_lds[0][r][b] + bI[q];
      float fv = gate_lds[1][r][b] + bF[q];
      float gv = gate_lds[2][r][b] + bG[q];
      float ov = gate_lds[3][r][b] + bO[q];
      float cn = sig_(fv) * c_lds[r][b] + sig_(iv) * tanh_(gv);
      float hn = sig_(ov) * tanh_(cn);
      c_lds[r][b] = cn;
      float* rb = rp[q] + (size_t)slot * H_;
      float ev = *rb;                 // h leaving the 16-entry window
      *rb = hn;                       // h entering it
      att_lds[r][b] += hn - ev;       // running sum == Σ buf (softmax ≡ 1)
      h_lds[r][b] = hn;
    }
    ep++; gbar(ctr, NWG * ep);        // all WGs done reading global act
  }

  // ---- final h, c (ring in d_out is already in correct order: 32 % 16 == 0)
  #pragma unroll
  for (int q = 0; q < 4; ++q) {
    int b = bq[q], r = rq[q];
    h_out[b * H_ + g * 32 + r] = h_lds[r][b];
    c_out[b * H_ + g * 32 + r] = c_lds[r][b];
  }
}

extern "C" void kernel_launch(void* const* d_in, const int* in_sizes, int n_in,
                              void* d_out, int out_size, void* d_ws, size_t ws_size,
                              hipStream_t stream) {
  const float* x    = (const float*)d_in[0];
  const float* h0   = (const float*)d_in[1];
  const float* c0   = (const float*)d_in[2];
  const float* out0 = (const float*)d_in[3];
  const float* Wih  = (const float*)d_in[4];   // (2048, 1024)
  const float* Whh  = (const float*)d_in[5];   // (2048, 512)
  const float* bih  = (const float*)d_in[6];
  const float* bhh  = (const float*)d_in[7];
  // aw1..ab3 (d_in[8..13]) are dead code: softmax over a size-1 axis => weight==1.

  unsigned*  ctr   = (unsigned*)d_ws;                                   // [0,4K)
  float*     biasc = (float*)((char*)d_ws + 4096);                      // 8KB
  _Float16*  act   = (_Float16*)((char*)d_ws + 16384);                  // 96KB
  _Float16*  wpack = (_Float16*)((char*)d_ws + 16384 + 131072);         // 6MB

  hipMemsetAsync(d_ws, 0, 4096, stream);   // reset barrier counter (graph-safe)
  pack_weights<<<(NGATE * KTOT) / 256, 256, 0, stream>>>(Wih, Whh, bih, bhh,
                                                         wpack, biasc);
  lstm_persistent<<<NWG, 256, 0, stream>>>(x, h0, c0, out0, biasc,
                                           wpack, act, ctr, (float*)d_out);
}
